// Attention_aware_msa_12103217840306
// MI455X (gfx1250) — matrix operand order repaired and checked by reading, not compiled, not run on hardware
//
#include <hip/hip_runtime.h>
#include <hip/hip_bf16.h>
#include <stdint.h>

#define NT 2048          // tokens
#define CD 1024          // channels
#define NH 8             // heads
#define HD 128           // head dim
#define C3 3072          // 3*C
#define SCALE 25.0f

typedef __attribute__((ext_vector_type(16))) __bf16 v16bf;
typedef __attribute__((ext_vector_type(8)))  float  v8f;
typedef __attribute__((ext_vector_type(4)))  unsigned int v4u;

union Frag16 { v16bf v; v4u q[2]; };

__device__ __forceinline__ unsigned short f2bf(float f){
  union { float f; unsigned int u; } x; x.f = f;
  unsigned int r = x.u + 0x7FFFu + ((x.u >> 16) & 1u);
  return (unsigned short)(r >> 16);
}

// A fragment 16x32 (MxK) bf16, src row-major [row][k] at (row0,k0), ld in elements.
// ISA layout: lanes 0-15 row=lane, elems 0..7 -> K 0..7, elems 8..15 -> K 16..23;
// lanes 16-31 same rows, K offset +8. Two 16B contiguous loads per lane.
__device__ __forceinline__ v16bf load_fragA(const unsigned short* src, int ld){
  const int lane = threadIdx.x & 31;
  const unsigned short* p = src + (lane & 15) * ld + ((lane >> 4) << 3);
  Frag16 f;
  f.q[0] = *(const v4u*)(p);
  f.q[1] = *(const v4u*)(p + 16);
  return f.v;
}

// B fragment 32x16 (KxN) bf16 for NT-GEMM: B[k][n] = src[n][k], src row-major, ld elems.
// lanes 0-15: n=lane, elems 0..7 -> K 0..7, elems 8..15 -> K 16..23; lanes 16-31: K offset +8.
__device__ __forceinline__ v16bf load_fragB(const unsigned short* src, int ld){
  const int lane = threadIdx.x & 31;
  const unsigned short* p = src + (lane & 15) * ld + ((lane >> 4) << 3);
  Frag16 f;
  f.q[0] = *(const v4u*)(p);
  f.q[1] = *(const v4u*)(p + 16);
  return f.v;
}

__device__ __forceinline__ v8f wmma_bf(v16bf a, v16bf b, v8f c){
  return __builtin_amdgcn_wmma_f32_16x16x32_bf16(false, a, false, b, (short)0, c,
                                                 false, false);
}

__device__ __forceinline__ v8f v8zero(){
  v8f z = {0.f,0.f,0.f,0.f,0.f,0.f,0.f,0.f};
  return z;
}

// ---------------------------------------------------------------- utility kernels
__global__ void k_zero(float* __restrict__ p, int n){
  int i = blockIdx.x * blockDim.x + threadIdx.x;
  if (i < n) p[i] = 0.f;
}

__global__ void k_f32_to_bf16(const float* __restrict__ s, unsigned short* __restrict__ d, int n){
  int i = blockIdx.x * blockDim.x + threadIdx.x;
  if (i < n) d[i] = f2bf(s[i]);
}

// ---------------------------------------------------------------- NT WMMA GEMM, 16x64/wave
// C[M][Nc] = scale * A[M][K] * B[Nc][K]^T  (A,B bf16 row-major, K contiguous).
// Block = 4 waves stacked on rows sharing one 64-column group (B frags hit L0).
// k-loop unrolled x2 with ping-pong fragment buffers: loads write each buffer
// directly (no rotation movs) and each WMMA group waits only on loads issued a
// full phase earlier -> VMEM fully overlapped with the matrix pipe.
__global__ void __launch_bounds__(128)
k_gemm_nt(const unsigned short* __restrict__ A, const unsigned short* __restrict__ B,
          float* __restrict__ C, int M, int Nc, int K, float scale){
  const int lane = threadIdx.x & 31;
  const int wid  = threadIdx.x >> 5;
  const int nbj  = Nc >> 6;                 // 64-column groups
  const int bi   = blockIdx.x / nbj;
  const int bj   = blockIdx.x % nbj;
  const int it   = (bi << 2) + wid;         // one 16-row tile per wave
  (void)M;
  const unsigned short* ap = A + (size_t)(it << 4) * K;
  const unsigned short* bp = B + (size_t)(bj << 6) * K;
  v8f acc[4];
#pragma unroll
  for (int j = 0; j < 4; ++j) acc[j] = v8zero();

  v16bf a0 = load_fragA(ap, K);
  v16bf b0[4];
#pragma unroll
  for (int j = 0; j < 4; ++j) b0[j] = load_fragB(bp + ((size_t)(j << 4)) * K, K);

  for (int k = 0; k < K; k += 64){
    // phase 0: prefetch k+32 into buffer 1, compute with buffer 0
    v16bf a1 = load_fragA(ap + k + 32, K);
    v16bf b1[4];
#pragma unroll
    for (int j = 0; j < 4; ++j) b1[j] = load_fragB(bp + ((size_t)(j << 4)) * K + k + 32, K);
#pragma unroll
    for (int j = 0; j < 4; ++j) acc[j] = wmma_bf(a0, b0[j], acc[j]);

    // phase 1: prefetch k+64 (wrap on last step) into buffer 0, compute with buffer 1
    const int kn = (k + 64 < K) ? k + 64 : 0;
    a0 = load_fragA(ap + kn, K);
#pragma unroll
    for (int j = 0; j < 4; ++j) b0[j] = load_fragB(bp + ((size_t)(j << 4)) * K + kn, K);
#pragma unroll
    for (int j = 0; j < 4; ++j) acc[j] = wmma_bf(a1, b1[j], acc[j]);
  }

  const int col = lane & 15, half = lane >> 4;
#pragma unroll
  for (int j = 0; j < 4; ++j){
    float* cp = C + (size_t)(it << 4) * Nc + (bj << 6) + (j << 4) + col;
#pragma unroll
    for (int i = 0; i < 8; ++i)
      cp[(size_t)(i + 8 * half) * Nc] = acc[j][i] * scale;
  }
}

// ---------------------------------------------------------------- normalize / scatter
// qkvf: [NT][3072] f32. Per (h,n) wave: L2 norms of q,k,v (128 each).
// Writes qn/kn [h][n][d] bf16, vT [h][d][n] bf16, vn [n][h*128+d] bf16,
// and x_ori half of output: outp[n*2048 + 1024 + h*128 + d] = v (f32).
__global__ void __launch_bounds__(256)
k_normalize(const float* __restrict__ qkvf,
            unsigned short* __restrict__ qn, unsigned short* __restrict__ kn,
            unsigned short* __restrict__ vT, unsigned short* __restrict__ vn,
            float* __restrict__ outp){
  const int w    = (blockIdx.x << 3) + (threadIdx.x >> 5);
  const int lane = threadIdx.x & 31;
  const int h = w >> 11, n = w & (NT - 1);
  if (h >= NH) return;
  const float* row = qkvf + (size_t)n * C3 + h * HD + lane * 4;
  float4 q = *(const float4*)(row);
  float4 k = *(const float4*)(row + CD);
  float4 v = *(const float4*)(row + 2 * CD);
  float sq = q.x*q.x + q.y*q.y + q.z*q.z + q.w*q.w;
  float sk = k.x*k.x + k.y*k.y + k.z*k.z + k.w*k.w;
  float sv = v.x*v.x + v.y*v.y + v.z*v.z + v.w*v.w;
#pragma unroll
  for (int o = 1; o < 32; o <<= 1){
    sq += __shfl_xor(sq, o);
    sk += __shfl_xor(sk, o);
    sv += __shfl_xor(sv, o);
  }
  const float rq = rsqrtf(sq), rk = rsqrtf(sk), rv = rsqrtf(sv);
  const int d0 = lane * 4;
  unsigned short* qp = qn + ((size_t)(h * NT + n)) * HD + d0;
  unsigned short* kp = kn + ((size_t)(h * NT + n)) * HD + d0;
  float qa[4] = {q.x,q.y,q.z,q.w}, ka[4] = {k.x,k.y,k.z,k.w}, va[4] = {v.x,v.y,v.z,v.w};
#pragma unroll
  for (int i = 0; i < 4; ++i){
    qp[i] = f2bf(qa[i] * rq);
    kp[i] = f2bf(ka[i] * rk);
    vT[((size_t)(h * HD + d0 + i)) * NT + n] = f2bf(va[i]);
    vn[(size_t)n * CD + h * HD + d0 + i]     = f2bf(va[i] * rv);
  }
  *(float4*)(outp + (size_t)n * (2 * CD) + CD + h * HD + d0) = v;
}

// ---------------------------------------------------------------- fused attention
// Block = 4 waves = 4 row-tiles of ONE head, sharing LDS-staged K/V tiles (4x less
// L2 traffic). Q fragments are loop-invariant: preloaded once, live across both
// sweeps. Two sweeps over 2048 keys (32/step):
//  sweep 1: row sums of exp(logits) (no max needed: |logit| <= 25), K staged in LDS;
//  sweep 2: P = 0.5*(softmax_cls+softmax_reg); atomicAdd P/8 into simAttn;
//           re-fragment P via per-wave LDS bounce; P@Vcls, P@Vreg WMMA from staged V.
__global__ void __launch_bounds__(128)
k_attention(const unsigned short* __restrict__ qn_c, const unsigned short* __restrict__ kn_c,
            const unsigned short* __restrict__ vT_c,
            const unsigned short* __restrict__ qn_r, const unsigned short* __restrict__ kn_r,
            const unsigned short* __restrict__ vT_r,
            const float* __restrict__ cls_score,
            float* __restrict__ simAttn,
            float* __restrict__ out_cls, float* __restrict__ out_reg){
  __shared__ alignas(16) unsigned short Kc_s[32 * HD];   // K tile: [m 0..31][d 0..127]
  __shared__ alignas(16) unsigned short Kr_s[32 * HD];
  __shared__ alignas(16) unsigned short Vc_s[HD * 32];   // V tile: [d 0..127][m 0..31]
  __shared__ alignas(16) unsigned short Vr_s[HD * 32];
  __shared__ alignas(16) unsigned short Pb[4][16 * 32];
  const int tid  = threadIdx.x;
  const int lane = tid & 31, wid = tid >> 5;
  const int h    = blockIdx.x >> 5;         // 32 blocks per head
  const int g    = blockIdx.x & 31;
  const int ntile = (g << 2) + wid;         // this wave's 16-row tile
  const int col = lane & 15, half = lane >> 4;

  const unsigned short* qc = qn_c + ((size_t)(h * NT + ntile * 16)) * HD;
  const unsigned short* qr = qn_r + ((size_t)(h * NT + ntile * 16)) * HD;
  const unsigned short* kc = kn_c + (size_t)h * NT * HD;
  const unsigned short* kr = kn_r + (size_t)h * NT * HD;
  const unsigned short* vc = vT_c + (size_t)h * HD * NT;
  const unsigned short* vr = vT_r + (size_t)h * HD * NT;

  // loop-invariant Q fragments (4 k-steps x 2 streams)
  v16bf qfc[4], qfr[4];
#pragma unroll
  for (int kk = 0; kk < 4; ++kk){
    qfc[kk] = load_fragA(qc + kk * 32, HD);
    qfr[kk] = load_fragA(qr + kk * 32, HD);
  }

  // ---- sweep 1: per-row exp sums (C-layout partials, one cross-lane reduce at end)
  float rsc[8], rsr[8];
#pragma unroll
  for (int i = 0; i < 8; ++i){ rsc[i] = 0.f; rsr[i] = 0.f; }

  for (int ms = 0; ms < 64; ++ms){
    const int m0 = ms * 32;
    {   // K tiles are contiguous 8KB blocks (ld == HD): straight block copy
      const v4u* sc_ = (const v4u*)(kc + (size_t)m0 * HD);
      const v4u* sr_ = (const v4u*)(kr + (size_t)m0 * HD);
      v4u* dc = (v4u*)Kc_s;
      v4u* dr = (v4u*)Kr_s;
#pragma unroll
      for (int c = 0; c < 4; ++c){
        dc[tid + c * 128] = sc_[tid + c * 128];
        dr[tid + c * 128] = sr_[tid + c * 128];
      }
    }
    __builtin_prefetch(kc + (size_t)(m0 + 32) * HD, 0, 0);   // global_prefetch_b8
    __builtin_prefetch(kr + (size_t)(m0 + 32) * HD, 0, 0);
    __syncthreads();
#pragma unroll
    for (int sub = 0; sub < 2; ++sub){
      v8f sc = v8zero(), sr = v8zero();
#pragma unroll
      for (int kk = 0; kk < 4; ++kk){
        sc = wmma_bf(qfc[kk], load_fragB(Kc_s + sub * 16 * HD + kk * 32, HD), sc);
        sr = wmma_bf(qfr[kk], load_fragB(Kr_s + sub * 16 * HD + kk * 32, HD), sr);
      }
      const float cs = cls_score[m0 + sub * 16 + col] * SCALE;
#pragma unroll
      for (int i = 0; i < 8; ++i){
        rsc[i] += __expf(sc[i] * cs);
        rsr[i] += __expf(sr[i] * SCALE);
      }
    }
    __syncthreads();
  }
#pragma unroll
  for (int i = 0; i < 8; ++i)
#pragma unroll
    for (int o = 1; o < 16; o <<= 1){
      rsc[i] += __shfl_xor(rsc[i], o);
      rsr[i] += __shfl_xor(rsr[i], o);
    }
  float inv_c[8], inv_r[8];
#pragma unroll
  for (int i = 0; i < 8; ++i){ inv_c[i] = 1.f / rsc[i]; inv_r[i] = 1.f / rsr[i]; }

  // ---- sweep 2: P, simAttn atomics, P@V
  v8f accC[8], accR[8];
#pragma unroll
  for (int t = 0; t < 8; ++t){ accC[t] = v8zero(); accR[t] = v8zero(); }
  unsigned short* P = &Pb[wid][0];

  for (int ms = 0; ms < 64; ++ms){
    const int m0 = ms * 32;
    {   // stage K (contiguous) and V (one 64B row slice per thread)
      const v4u* sc_ = (const v4u*)(kc + (size_t)m0 * HD);
      const v4u* sr_ = (const v4u*)(kr + (size_t)m0 * HD);
      v4u* dc = (v4u*)Kc_s;
      v4u* dr = (v4u*)Kr_s;
#pragma unroll
      for (int c = 0; c < 4; ++c){
        dc[tid + c * 128] = sc_[tid + c * 128];
        dr[tid + c * 128] = sr_[tid + c * 128];
      }
      const v4u* s1 = (const v4u*)(vc + (size_t)tid * NT + m0);
      const v4u* s2 = (const v4u*)(vr + (size_t)tid * NT + m0);
      v4u* d1 = (v4u*)(Vc_s + tid * 32);
      v4u* d2 = (v4u*)(Vr_s + tid * 32);
#pragma unroll
      for (int c = 0; c < 4; ++c){ d1[c] = s1[c]; d2[c] = s2[c]; }
    }
    __builtin_prefetch(kc + (size_t)(m0 + 32) * HD, 0, 0);
    __builtin_prefetch(kr + (size_t)(m0 + 32) * HD, 0, 0);
    __syncthreads();
#pragma unroll
    for (int sub = 0; sub < 2; ++sub){
      v8f sc = v8zero(), sr = v8zero();
#pragma unroll
      for (int kk = 0; kk < 4; ++kk){
        sc = wmma_bf(qfc[kk], load_fragB(Kc_s + sub * 16 * HD + kk * 32, HD), sc);
        sr = wmma_bf(qfr[kk], load_fragB(Kr_s + sub * 16 * HD + kk * 32, HD), sr);
      }
      const float cs = cls_score[m0 + sub * 16 + col] * SCALE;
      const int mg = m0 + sub * 16 + col;
#pragma unroll
      for (int i = 0; i < 8; ++i){
        const int row = i + 8 * half;
        const float p = 0.5f * (__expf(sc[i] * cs)    * inv_c[i] +
                                __expf(sr[i] * SCALE) * inv_r[i]);
        atomicAdd(&simAttn[(size_t)(ntile * 16 + row) * NT + mg], p * 0.125f);
        P[row * 32 + sub * 16 + col] = f2bf(p);
      }
    }
    const v16bf pa = load_fragA(P, 32);  // C-layout -> A-layout via per-wave LDS (in order)
#pragma unroll
    for (int t = 0; t < 8; ++t){
      accC[t] = wmma_bf(pa, load_fragB(Vc_s + t * 16 * 32, 32), accC[t]);
      accR[t] = wmma_bf(pa, load_fragB(Vr_s + t * 16 * 32, 32), accR[t]);
    }
    __syncthreads();
  }

  float* oc  = out_cls + (size_t)(ntile * 16) * (2 * CD) + h * HD;
  float* orr = out_reg + (size_t)(ntile * 16) * (2 * CD) + h * HD;
#pragma unroll
  for (int t = 0; t < 8; ++t)
#pragma unroll
    for (int i = 0; i < 8; ++i){
      const int row = i + 8 * half;
      oc [(size_t)row * (2 * CD) + t * 16 + col] = accC[t][i];
      orr[(size_t)row * (2 * CD) + t * 16 + col] = accR[t][i];
    }
}

// ---------------------------------------------------------------- final masked renorm
// sr2 = mask_sim*exp(sim) / sum(mask_sim*exp(sim))   (softmax normalizer cancels)
// obj = mask_obj*mask_sim*exp(sim) / sum(mask_obj*mask_sim*exp(sim))
__global__ void __launch_bounds__(256)
k_final(const float* __restrict__ simAttn, const float* __restrict__ acr,
        const float* __restrict__ arr, float* __restrict__ sr2, float* __restrict__ obj){
  __shared__ float red[256];
  const int n = blockIdx.x, t = threadIdx.x;
  const float* srow = simAttn + (size_t)n * NT;
  const float* ar   = acr + (size_t)n * NT;
  const float* br   = arr + (size_t)n * NT;
  float a[8], ba[8];
  float sa = 0.f, sb = 0.f;
#pragma unroll
  for (int j = 0; j < 8; ++j){
    const int m = t + j * 256;
    const float ev = __expf(srow[m]);
    const float av = (ar[m] > 0.75f) ? ev : 0.f;
    const float bv = (br[m] > 0.99f) ? av : 0.f;
    a[j] = av; ba[j] = bv; sa += av; sb += bv;
  }
  red[t] = sa; __syncthreads();
  for (int o = 128; o > 0; o >>= 1){ if (t < o) red[t] += red[t + o]; __syncthreads(); }
  sa = red[0]; __syncthreads();
  red[t] = sb; __syncthreads();
  for (int o = 128; o > 0; o >>= 1){ if (t < o) red[t] += red[t + o]; __syncthreads(); }
  sb = red[0];
  const float isa = 1.f / sa, isb = 1.f / sb;
#pragma unroll
  for (int j = 0; j < 8; ++j){
    const int m = t + j * 256;
    sr2[(size_t)n * NT + m] = a[j] * isa;
    obj[(size_t)n * NT + m] = ba[j] * isb;
  }
}

// ---------------------------------------------------------------- launcher
extern "C" void kernel_launch(void* const* d_in, const int* in_sizes, int n_in,
                              void* d_out, int out_size, void* d_ws, size_t ws_size,
                              hipStream_t stream){
  (void)in_sizes; (void)n_in; (void)out_size; (void)ws_size;
  const float* x_cls = (const float*)d_in[0];
  const float* x_reg = (const float*)d_in[1];
  const float* cls_s = (const float*)d_in[2];
  // d_in[3] = fg_score: unused by reference
  const float* W_cls = (const float*)d_in[4];
  const float* W_reg = (const float*)d_in[5];

  float* out     = (float*)d_out;
  float* out_cls = out;
  float* out_reg = out + (size_t)NT * 2 * CD;
  float* sr2     = out + (size_t)2 * NT * 2 * CD;
  float* obj     = out + (size_t)3 * NT * 2 * CD;

  char* wp = (char*)d_ws;
  auto alloc = [&](size_t bytes) -> char* {
    char* p = wp; wp += (bytes + 255) & ~(size_t)255; return p;
  };
  unsigned short* x16c = (unsigned short*)alloc((size_t)NT * CD * 2);
  unsigned short* x16r = (unsigned short*)alloc((size_t)NT * CD * 2);
  unsigned short* w16c = (unsigned short*)alloc((size_t)C3 * CD * 2);
  unsigned short* w16r = (unsigned short*)alloc((size_t)C3 * CD * 2);
  float*          qkvf = (float*)         alloc((size_t)NT * C3 * 4);   // reused cls->reg
  unsigned short* qn_c = (unsigned short*)alloc((size_t)NH * NT * HD * 2);
  unsigned short* kn_c = (unsigned short*)alloc((size_t)NH * NT * HD * 2);
  unsigned short* vT_c = (unsigned short*)alloc((size_t)NH * HD * NT * 2);
  unsigned short* vn_c = (unsigned short*)alloc((size_t)NT * CD * 2);
  unsigned short* qn_r = (unsigned short*)alloc((size_t)NH * NT * HD * 2);
  unsigned short* kn_r = (unsigned short*)alloc((size_t)NH * NT * HD * 2);
  unsigned short* vT_r = (unsigned short*)alloc((size_t)NH * HD * NT * 2);
  unsigned short* vn_r = (unsigned short*)alloc((size_t)NT * CD * 2);
  float*          simA = (float*)         alloc((size_t)NT * NT * 4);
  float*          acr  = (float*)         alloc((size_t)NT * NT * 4);
  float*          arr  = (float*)         alloc((size_t)NT * NT * 4);

  // 0) zero the head-averaged attention accumulator
  k_zero<<<(NT * NT + 255) / 256, 256, 0, stream>>>(simA, NT * NT);

  // 1) f32 -> bf16 conversions
  k_f32_to_bf16<<<(NT * CD + 255) / 256, 256, 0, stream>>>(x_cls, x16c, NT * CD);
  k_f32_to_bf16<<<(NT * CD + 255) / 256, 256, 0, stream>>>(x_reg, x16r, NT * CD);
  k_f32_to_bf16<<<(C3 * CD + 255) / 256, 256, 0, stream>>>(W_cls, w16c, C3 * CD);
  k_f32_to_bf16<<<(C3 * CD + 255) / 256, 256, 0, stream>>>(W_reg, w16r, C3 * CD);

  // 2) QKV GEMM + normalize, cls then reg (qkvf reused)
  const int qkv_blocks = (NT / 64) * (C3 / 64);       // 16x64 per wave, 4 waves/block
  k_gemm_nt<<<qkv_blocks, 128, 0, stream>>>(x16c, w16c, qkvf, NT, C3, CD, 1.0f);
  k_normalize<<<NH * NT / 8, 256, 0, stream>>>(qkvf, qn_c, kn_c, vT_c, vn_c, out_cls);
  k_gemm_nt<<<qkv_blocks, 128, 0, stream>>>(x16r, w16r, qkvf, NT, C3, CD, 1.0f);
  k_normalize<<<NH * NT / 8, 256, 0, stream>>>(qkvf, qn_r, kn_r, vT_r, vn_r, out_reg);

  // 3) fused dual-stream attention: 8 heads x 32 blocks, 4 row-tiles/block
  k_attention<<<NH * 32, 128, 0, stream>>>(qn_c, kn_c, vT_c, qn_r, kn_r, vT_r,
                                           cls_s, simA, out_cls, out_reg);

  // 4) head-averaged Vn*Vn^T similarity matrices (scale 1/H)
  const int sim_blocks = (NT / 64) * (NT / 64);
  k_gemm_nt<<<sim_blocks, 128, 0, stream>>>(vn_c, vn_c, acr, NT, NT, CD, 0.125f);
  k_gemm_nt<<<sim_blocks, 128, 0, stream>>>(vn_r, vn_r, arr, NT, NT, CD, 0.125f);

  // 5) masked renormalizations -> sim_round2, obj_mask
  k_final<<<NT, 256, 0, stream>>>(simA, acr, arr, sr2, obj);
}
